// test_PAConv_72919954752270
// MI455X (gfx1250) — matrix operand order repaired and checked by reading, not compiled, not run on hardware
//
#include <hip/hip_runtime.h>
#include <hip/hip_bf16.h>

typedef __attribute__((ext_vector_type(16))) __bf16 v16bf;
typedef __attribute__((ext_vector_type(8)))  float  v8f;

#define BATCH 4
#define NPTS  1024
#define KNN   30
#define MNUM  8
#define ODIM  64
#define OUTD  512
#define FEATC 320
static __device__ __constant__ float BN_SCALE_C = 0.99999500003749981f; // 1/sqrt(1+1e-5)

// ---------------------------------------------------------------------------
// 0) transpose x (B,9,N) -> xt (B,N,9)
// ---------------------------------------------------------------------------
__global__ void k_transpose_x(const float* __restrict__ x, float* __restrict__ xt) {
    int i = blockIdx.x * 256 + threadIdx.x;
    if (i >= BATCH * NPTS * 9) return;
    int b = i / (NPTS * 9);
    int r = i - b * (NPTS * 9);
    int n = r / 9, c = r - n * 9;
    xt[i] = x[(b * 9 + c) * NPTS + n];
}

// ---------------------------------------------------------------------------
// 1) kNN: one block (256 thr) per point; top-30 of -||xi-xj||^2, index tiebreak
// ---------------------------------------------------------------------------
__global__ void k_knn(const float* __restrict__ xt, int* __restrict__ idxout) {
    int bn = blockIdx.x;
    int b = bn >> 10, n = bn & (NPTS - 1);
    __shared__ float dist[NPTS];
    __shared__ float rval[256];
    __shared__ int   ridx[256];
    int t = threadIdx.x;
    const float* base = xt + (size_t)b * NPTS * 9;
    float ctr[9];
    const float* cp = base + n * 9;
#pragma unroll
    for (int c = 0; c < 9; ++c) ctr[c] = cp[c];
    for (int j = t; j < NPTS; j += 256) {
        const float* pj = base + j * 9;
        float d = 0.f;
#pragma unroll
        for (int c = 0; c < 9; ++c) { float df = pj[c] - ctr[c]; d += df * df; }
        dist[j] = -d;
    }
    __syncthreads();
    for (int k = 0; k < KNN; ++k) {
        float bv = -3.4e38f; int bi = 0x7fffffff;
        for (int j = t; j < NPTS; j += 256) {
            float v = dist[j];
            if (v > bv || (v == bv && j < bi)) { bv = v; bi = j; }
        }
        rval[t] = bv; ridx[t] = bi;
        __syncthreads();
        for (int s = 128; s > 0; s >>= 1) {
            if (t < s) {
                float v2 = rval[t + s]; int i2 = ridx[t + s];
                if (v2 > rval[t] || (v2 == rval[t] && i2 < ridx[t])) { rval[t] = v2; ridx[t] = i2; }
            }
            __syncthreads();
        }
        if (t == 0) { int w = ridx[0]; idxout[bn * KNN + k] = w; dist[w] = -3.4e38f; }
        __syncthreads();
    }
}

// ---------------------------------------------------------------------------
// 2) per-point geometry, conv1(+bn,relu,max_k) -> x1, and scorenet for all 4
//    layers -> score buffer. One block (256 thr) per point.
// ---------------------------------------------------------------------------
__global__ void k_point_features(const float* __restrict__ xt, const int* __restrict__ idx,
                                 const float* __restrict__ w1, const float* __restrict__ b1,
                                 const float* __restrict__ hw, const float* __restrict__ ow,
                                 const float* __restrict__ ob,
                                 float* __restrict__ xx, __bf16* __restrict__ fbf,
                                 float* __restrict__ scoreBuf) {
    int bn = blockIdx.x;
    int b = bn >> 10, n = bn & (NPTS - 1);
    int t = threadIdx.x;
    __shared__ float ctr[9];
    __shared__ int   knn[KNN];
    __shared__ float xyz[KNN][28];
    __shared__ float gfm[KNN][18];
    __shared__ float hbuf[KNN][ODIM];
    __shared__ float hid[KNN][16];
    __shared__ float sco[KNN][MNUM];
    const float* base = xt + (size_t)b * NPTS * 9;
    if (t < 9)   ctr[t] = base[n * 9 + t];
    if (t < KNN) knn[t] = idx[bn * KNN + t];
    __syncthreads();
    for (int u = t; u < KNN * 9; u += 256) {
        int k = u / 9, c = u - k * 9;
        float nb = base[knn[k] * 9 + c];
        float df = nb - ctr[c];
        xyz[k][c] = df; xyz[k][9 + c] = nb; xyz[k][18 + c] = ctr[c];
        gfm[k][c] = df; gfm[k][9 + c] = ctr[c];
    }
    __syncthreads();
    if (t < KNN) {
        float s = 0.f;
#pragma unroll
        for (int c = 0; c < 9; ++c) { float d = xyz[t][c]; s += d * d; }
        xyz[t][27] = sqrtf(s);
    }
    __syncthreads();
    // conv1 -> bn -> relu
    for (int u = t; u < KNN * ODIM; u += 256) {
        int k = u >> 6, o = u & 63;
        float acc = b1[o];
        const float* wr = w1 + o * 18;
#pragma unroll
        for (int c = 0; c < 18; ++c) acc += gfm[k][c] * wr[c];
        acc *= BN_SCALE_C;
        hbuf[k][o] = acc > 0.f ? acc : 0.f;
    }
    __syncthreads();
    if (t < ODIM) {
        float mx = hbuf[0][t];
        for (int k = 1; k < KNN; ++k) mx = fmaxf(mx, hbuf[k][t]);
        xx[(size_t)bn * FEATC + t] = mx;
        fbf[(size_t)bn * ODIM + t] = (__bf16)mx;
    }
    // scorenet for all 4 layers (xyz is layer-invariant)
    for (int l = 0; l < 4; ++l) {
        const float* hwl = hw + l * 16 * 28;
        const float* owl = ow + l * MNUM * 16;
        const float* obl = ob + l * MNUM;
        __syncthreads();
        for (int u = t; u < KNN * 16; u += 256) {
            int k = u >> 4, hh = u & 15;
            float acc = 0.f;
            const float* wr = hwl + hh * 28;
#pragma unroll
            for (int c = 0; c < 28; ++c) acc += xyz[k][c] * wr[c];
            acc *= BN_SCALE_C;
            hid[k][hh] = acc > 0.f ? acc : 0.f;
        }
        __syncthreads();
        for (int u = t; u < KNN * MNUM; u += 256) {
            int k = u >> 3, m = u & 7;
            float acc = obl[m];
            const float* wr = owl + m * 16;
#pragma unroll
            for (int c = 0; c < 16; ++c) acc += hid[k][c] * wr[c];
            sco[k][m] = acc;
        }
        __syncthreads();
        if (t < KNN) {
            float mx = sco[t][0];
#pragma unroll
            for (int m = 1; m < MNUM; ++m) mx = fmaxf(mx, sco[t][m]);
            float e[MNUM]; float s = 0.f;
#pragma unroll
            for (int m = 0; m < MNUM; ++m) { e[m] = expf(sco[t][m] - mx); s += e[m]; }
            float inv = 1.f / s;
            float* dst = scoreBuf + ((size_t)l * (BATCH * NPTS) + bn) * (KNN * MNUM) + t * MNUM;
#pragma unroll
            for (int m = 0; m < MNUM; ++m) dst[m] = e[m] * inv;
        }
    }
}

// ---------------------------------------------------------------------------
// 3) weight prep: ktopT / ksumT (N-major bf16, n*64+k) from mats (4,128,512)
// ---------------------------------------------------------------------------
__global__ void k_prep_kern(const float* __restrict__ mats, __bf16* __restrict__ ktopT,
                            __bf16* __restrict__ ksumT) {
    int i = blockIdx.x * 256 + threadIdx.x;
    if (i >= 4 * OUTD * ODIM) return;
    int l = i >> 15, r = i & 32767;
    int n = r >> 6, k = r & 63;
    const float* m = mats + (size_t)l * 128 * OUTD;
    float top = m[k * OUTD + n];
    float bot = m[(k + ODIM) * OUTD + n];
    ktopT[i] = (__bf16)top;
    ksumT[i] = (__bf16)(top + bot);
}

__global__ void k_cvt_bf16(const float* __restrict__ in, __bf16* __restrict__ out, int nElem) {
    int i = blockIdx.x * 256 + threadIdx.x;
    if (i < nElem) out[i] = (__bf16)in[i];
}

// ---------------------------------------------------------------------------
// 4) bf16 WMMA GEMM: C(M,N) = A(M,K) @ W(K,N), W supplied N-major (WT[n*K+k]).
//    One wave per 16x16 tile, K stepped by 32 through v_wmma_f32_16x16x32_bf16.
// ---------------------------------------------------------------------------
__global__ void k_gemm_bf16_wmma(const __bf16* __restrict__ A, const __bf16* __restrict__ WT,
                                 float* __restrict__ C, int Mdim, int Kdim, int Ndim,
                                 int reluBn) {
    int wave = blockIdx.x * (blockDim.x >> 5) + (threadIdx.x >> 5);
    int lane = threadIdx.x & 31;
    int tilesN = Ndim >> 4;
    int nt = (Mdim >> 4) * tilesN;
    if (wave >= nt) return;
    int tm = wave / tilesN, tn = wave - tm * tilesN;
    int l16  = lane & 15;
    int half = lane >> 4;
    int rowA = tm * 16 + l16;
    int coln = tn * 16 + l16;
    v8f acc = {};
    for (int k0 = 0; k0 < Kdim; k0 += 32) {
        v16bf a, bfr;
        // A 16x32 fragment: lane<16 holds K {0..15}; lane>=16 holds {16..31}
        const __bf16* ap = A + (size_t)rowA * Kdim + k0 + half * 16;
#pragma unroll
        for (int e = 0; e < 8; ++e)  a[e] = ap[e];
#pragma unroll
        for (int e = 8; e < 16; ++e) a[e] = ap[e];
        // B 32x16 fragment: lane holds column 'coln', K sequential (+16 for upper half)
        const __bf16* bp = WT + (size_t)coln * Kdim + k0 + half * 16;
#pragma unroll
        for (int e = 0; e < 16; ++e) bfr[e] = bp[e];
        acc = __builtin_amdgcn_wmma_f32_16x16x32_bf16(false, a, false, bfr,
                                                      (short)0, acc, false, false);
    }
#pragma unroll
    for (int r = 0; r < 8; ++r) {
        int m = tm * 16 + r + half * 8;
        float v = acc[r];
        if (reluBn) { v *= BN_SCALE_C; v = v > 0.f ? v : 0.f; }
        C[(size_t)m * Ndim + tn * 16 + l16] = v;
    }
}

// ---------------------------------------------------------------------------
// 5) assemble: out[bn,o] = sum_k sum_m sc*point[nbr] - sum_m ssum*center; relu(bn)
// ---------------------------------------------------------------------------
__global__ void k_assemble(const float* __restrict__ point, const float* __restrict__ center,
                           const float* __restrict__ scoreL, const int* __restrict__ idx,
                           float* __restrict__ xx, __bf16* __restrict__ fbf, int l) {
    int bn = blockIdx.x;
    int b = bn >> 10;
    int o = threadIdx.x; // 64
    __shared__ float sc[KNN][MNUM];
    __shared__ int   knn[KNN];
    __shared__ float ssum[MNUM];
    if (o < KNN) knn[o] = idx[bn * KNN + o];
    const float* srow = scoreL + (size_t)bn * (KNN * MNUM);
    for (int u = o; u < KNN * MNUM; u += 64) ((float*)sc)[u] = srow[u];
    __syncthreads();
    if (o < MNUM) {
        float s = 0.f;
        for (int k = 0; k < KNN; ++k) s += sc[k][o];
        ssum[o] = s;
    }
    __syncthreads();
    float acc = 0.f;
    for (int k = 0; k < KNN; ++k) {
        const float* prow = point + ((size_t)(b * NPTS + knn[k])) * OUTD;
#pragma unroll
        for (int m = 0; m < MNUM; ++m) acc += sc[k][m] * prow[m * ODIM + o];
    }
    const float* crow = center + (size_t)bn * OUTD;
#pragma unroll
    for (int m = 0; m < MNUM; ++m) acc -= ssum[m] * crow[m * ODIM + o];
    acc *= BN_SCALE_C;
    acc = acc > 0.f ? acc : 0.f;
    xx[(size_t)bn * FEATC + (l + 1) * ODIM + o] = acc;
    fbf[(size_t)bn * ODIM + o] = (__bf16)acc;
}

// ---------------------------------------------------------------------------
// 6) global max over N -> (B,512)
// ---------------------------------------------------------------------------
__global__ void k_max_over_n(const float* __restrict__ xc, float* __restrict__ out) {
    int i = blockIdx.x * 256 + threadIdx.x;
    if (i >= BATCH * OUTD) return;
    int b = i >> 9, o = i & (OUTD - 1);
    const float* p = xc + (size_t)b * NPTS * OUTD + o;
    float mx = -3.4e38f;
    for (int n = 0; n < NPTS; ++n) mx = fmaxf(mx, p[(size_t)n * OUTD]);
    out[i] = mx;
}

// ---------------------------------------------------------------------------
extern "C" void kernel_launch(void* const* d_in, const int* in_sizes, int n_in,
                              void* d_out, int out_size, void* d_ws, size_t ws_size,
                              hipStream_t stream) {
    const float* x        = (const float*)d_in[0];
    const float* conv1_w  = (const float*)d_in[3];
    const float* conv1_b  = (const float*)d_in[4];
    const float* sn_hw    = (const float*)d_in[5];
    const float* sn_ow    = (const float*)d_in[6];
    const float* sn_ob    = (const float*)d_in[7];
    const float* mats     = (const float*)d_in[8];
    const float* convt_w  = (const float*)d_in[9];
    float* out = (float*)d_out;

    // workspace carve-up (256B aligned)
    char* p = (char*)d_ws;
    auto carve = [&](size_t bytes) { char* r = p; p += (bytes + 255) & ~size_t(255); return (void*)r; };
    const int BN = BATCH * NPTS; // 4096
    float*  xt     = (float*) carve((size_t)BN * 9 * 4);
    int*    idx    = (int*)   carve((size_t)BN * KNN * 4);
    float*  score  = (float*) carve((size_t)4 * BN * KNN * MNUM * 4);
    float*  xx     = (float*) carve((size_t)BN * FEATC * 4);
    __bf16* fbf    = (__bf16*)carve((size_t)BN * ODIM * 2);
    __bf16* ktopT  = (__bf16*)carve((size_t)4 * OUTD * ODIM * 2);
    __bf16* ksumT  = (__bf16*)carve((size_t)4 * OUTD * ODIM * 2);
    __bf16* cwbf   = (__bf16*)carve((size_t)OUTD * FEATC * 2);
    __bf16* xxbf   = (__bf16*)carve((size_t)BN * FEATC * 2);
    float*  point  = (float*) carve((size_t)BN * OUTD * 4);
    float*  center = (float*) carve((size_t)BN * OUTD * 4);
    float*  xc     = (float*) carve((size_t)BN * OUTD * 4);

    // 0) transpose
    k_transpose_x<<<(BN * 9 + 255) / 256, 256, 0, stream>>>(x, xt);
    // 1) kNN
    k_knn<<<BN, 256, 0, stream>>>(xt, idx);
    // prep weights
    k_prep_kern<<<(4 * OUTD * ODIM + 255) / 256, 256, 0, stream>>>(mats, ktopT, ksumT);
    k_cvt_bf16<<<(OUTD * FEATC + 255) / 256, 256, 0, stream>>>(convt_w, cwbf, OUTD * FEATC);
    // 2) geometry + conv1 + scores
    k_point_features<<<BN, 256, 0, stream>>>(xt, idx, conv1_w, conv1_b, sn_hw, sn_ow, sn_ob,
                                             xx, fbf, score);
    // 3..6) PAConv layers: two WMMA GEMMs + assemble each
    int tiles = (BN / 16) * (OUTD / 16);        // 8192 tiles
    int gemmBlocks = (tiles + 3) / 4;           // 4 waves / 128-thread block
    for (int l = 0; l < 4; ++l) {
        k_gemm_bf16_wmma<<<gemmBlocks, 128, 0, stream>>>(fbf, ksumT + (size_t)l * OUTD * ODIM,
                                                         point, BN, ODIM, OUTD, 0);
        k_gemm_bf16_wmma<<<gemmBlocks, 128, 0, stream>>>(fbf, ktopT + (size_t)l * OUTD * ODIM,
                                                         center, BN, ODIM, OUTD, 0);
        k_assemble<<<BN, 64, 0, stream>>>(point, center,
                                          score + (size_t)l * BN * KNN * MNUM, idx,
                                          xx, fbf, l);
    }
    // 7) final conv (WMMA GEMM, K=320) with relu(bn) epilogue, then max over N
    k_cvt_bf16<<<(BN * FEATC + 255) / 256, 256, 0, stream>>>(xx, xxbf, BN * FEATC);
    k_gemm_bf16_wmma<<<gemmBlocks, 128, 0, stream>>>(xxbf, cwbf, xc, BN, FEATC, OUTD, 1);
    k_max_over_n<<<(BATCH * OUTD + 255) / 256, 256, 0, stream>>>(xc, out);
}
